// GUTF_47802986004832
// MI455X (gfx1250) — matrix operand order repaired and checked by reading, not compiled, not run on hardware
//
#include <hip/hip_runtime.h>
#include <hip/hip_bf16.h>
#include <cstddef>

// ---------------------------------------------------------------------------
// Problem constants (match reference setup_inputs)
// ---------------------------------------------------------------------------
#define BATCH 8
#define NDIM  2048
#define EDIM  8192
#define FDIM  64
#define NUM_HIDDEN 8

typedef __attribute__((ext_vector_type(16))) __bf16 v16bf;
typedef __attribute__((ext_vector_type(8)))  __bf16 v8bf;
typedef __attribute__((ext_vector_type(4)))  __bf16 v4bf;
typedef __attribute__((ext_vector_type(8)))  float  v8f;

// Epilogue modes
#define MODE_PLAIN 0
#define MODE_SOFT  1
#define MODE_ADDBX 2

// Tile config: 256 threads = 8 wave32; block computes 128 (M) x 64 (F) of C.
#define MTILE 128
#define KSTEP 32
#define LDS_PAD 8                    // row stride 32+8=40 elems = 80B (16B aligned)
#define LDSROW (KSTEP + LDS_PAD)

// ---------------------------------------------------------------------------
// One-time element-wise f32 -> bf16 conversion (grid-stride over float4)
// ---------------------------------------------------------------------------
__global__ __launch_bounds__(256)
void cvt_f32_to_bf16(const float* __restrict__ in, __bf16* __restrict__ out,
                     size_t n4)
{
    size_t i      = (size_t)blockIdx.x * 256 + threadIdx.x;
    size_t stride = (size_t)gridDim.x * 256;
    for (; i < n4; i += stride) {
        const float4 v = ((const float4*)in)[i];
        v4bf o;
        o[0] = (__bf16)v.x; o[1] = (__bf16)v.y;
        o[2] = (__bf16)v.z; o[3] = (__bf16)v.w;
        ((v4bf*)out)[i] = o;
    }
}

// ---------------------------------------------------------------------------
// One-time L [N,E] f32 -> Lbf [N,E] bf16  AND  LTbf [E,N] bf16 (transposed).
// 32x32 tile through LDS so both global writes stay coalesced.
// ---------------------------------------------------------------------------
__global__ __launch_bounds__(256)
void cvt_transpose_L(const float* __restrict__ L,
                     __bf16* __restrict__ Lbf,
                     __bf16* __restrict__ LTbf)
{
    __shared__ float t[32][33];
    const int tid = threadIdx.x;
    const int e0  = blockIdx.x * 32;   // column (E) tile
    const int n0  = blockIdx.y * 32;   // row (N) tile

    #pragma unroll
    for (int j = 0; j < 4; ++j) {
        const int idx = tid + j * 256;
        const int r = idx >> 5, c = idx & 31;          // r: n-local, c: e-local
        const float v = L[(size_t)(n0 + r) * EDIM + e0 + c];
        Lbf[(size_t)(n0 + r) * EDIM + e0 + c] = (__bf16)v;
        t[r][c] = v;
    }
    __syncthreads();
    #pragma unroll
    for (int j = 0; j < 4; ++j) {
        const int idx = tid + j * 256;
        const int r = idx >> 5, c = idx & 31;          // r: e-local, c: n-local
        LTbf[(size_t)(e0 + r) * NDIM + n0 + c] = (__bf16)t[c][r];
    }
}

// ---------------------------------------------------------------------------
// Generic tall-skinny GEMM:  C[M,64] = A[M,K] * B[K,64]  (+ epilogue)
// A always row-major (transposed operand is pre-materialized).
// TA/TB: float or __bf16 element type of A/B; TO: output element type.
// bf16 WMMA 16x16x32, f32 accumulation.
// ---------------------------------------------------------------------------
template <typename TA, typename TB, typename TO, int MODE>
__global__ __launch_bounds__(256)
void gemm_wmma(const TA* __restrict__ A, long lda, long strideA,
               const TB* __restrict__ Bm, long strideB,
               TO* __restrict__ out, long strideOut,
               const float* __restrict__ bx, long strideBx,
               const float* __restrict__ alphaPtr,
               int M, int K)
{
    constexpr bool A_F32 = (sizeof(TA) == 4);
    constexpr bool B_F32 = (sizeof(TB) == 4);

    __shared__ __bf16 Alds[MTILE * LDSROW];   // [128][40]
    __shared__ __bf16 Blds[FDIM  * LDSROW];   // [64][40]  (B staged transposed: [f][k])

    const int tid   = threadIdx.x;
    const int wave  = tid >> 5;               // 0..7 -> 16-row strip
    const int lane  = tid & 31;
    const int lrow  = lane & 15;              // A row / B column within fragment
    const int h     = lane >> 4;              // K-half selector (ISA 7.12.2 layout)

    const int batch = blockIdx.y;
    const int row0  = blockIdx.x * MTILE;

    A   += (size_t)batch * strideA;
    Bm  += (size_t)batch * strideB;
    out += (size_t)batch * strideOut;
    if (MODE == MODE_ADDBX) bx += (size_t)batch * strideBx;

    v8f acc[4];
    #pragma unroll
    for (int f = 0; f < 4; ++f)
        #pragma unroll
        for (int v = 0; v < 8; ++v) acc[f][v] = 0.0f;

    for (int k0 = 0; k0 < K; k0 += KSTEP) {
        __syncthreads();

        // ---- stage A tile: 128 x 32 -> bf16 LDS ----
        if constexpr (A_F32) {
            // f32 source: float4 loads + convert, 4 vec4 per thread
            #pragma unroll
            for (int j = 0; j < 4; ++j) {
                const int e  = (tid + j * 256) << 2;    // element id in 128x32
                const int m  = e >> 5;
                const int kk = e & 31;
                const float4 v = *(const float4*)(A + (size_t)(row0 + m) * lda + k0 + kk);
                __bf16* dst = &Alds[m * LDSROW + kk];
                dst[0] = (__bf16)v.x; dst[1] = (__bf16)v.y;
                dst[2] = (__bf16)v.z; dst[3] = (__bf16)v.w;
            }
        } else {
            // bf16 source: straight b128 copies, 2 x 8-elem chunks per thread
            #pragma unroll
            for (int j = 0; j < 2; ++j) {
                const int e  = (tid + j * 256) << 3;    // element id in 128x32
                const int m  = e >> 5;
                const int kk = e & 31;
                *(v8bf*)(&Alds[m * LDSROW + kk]) =
                    *(const v8bf*)(A + (size_t)(row0 + m) * lda + k0 + kk);
            }
        }

        // ---- stage B tile: 32 x 64 -> bf16 LDS, transposed to [f][k] ----
        #pragma unroll
        for (int j = 0; j < 8; ++j) {
            const int e  = tid + j * 256;               // 0..2047
            const int f  = e & 63;
            const int kk = e >> 6;
            if constexpr (B_F32) {
                Blds[f * LDSROW + kk] = (__bf16)Bm[(size_t)(k0 + kk) * FDIM + f];
            } else {
                Blds[f * LDSROW + kk] = Bm[(size_t)(k0 + kk) * FDIM + f];
            }
        }

        __syncthreads();

        // ---- A fragment (16x32 bf16, ISA 7.12.2 lane layout) ----
        const __bf16* arow = &Alds[(wave * 16 + lrow) * LDSROW];
        const v8bf alo = *(const v8bf*)(arow + h * 8);        // K {0..7} / {8..15}
        const v8bf ahi = *(const v8bf*)(arow + 16 + h * 8);   // K {16..23} / {24..31}
        v16bf afrag;
        #pragma unroll
        for (int i = 0; i < 8; ++i) { afrag[i] = alo[i]; afrag[i + 8] = ahi[i]; }

        // ---- 4 WMMAs across the 64-wide F dimension ----
        #pragma unroll
        for (int f = 0; f < 4; ++f) {
            const __bf16* bcol = &Blds[(f * 16 + lrow) * LDSROW + h * 8];
            const v8bf blo = *(const v8bf*)(bcol);
            const v8bf bhi = *(const v8bf*)(bcol + 16);
            v16bf bfrag;
            #pragma unroll
            for (int i = 0; i < 8; ++i) { bfrag[i] = blo[i]; bfrag[i + 8] = bhi[i]; }

            acc[f] = __builtin_amdgcn_wmma_f32_16x16x32_bf16(
                         false, afrag, false, bfrag,
                         (short)0, acc[f], false, false);
        }
    }

    // ---- epilogue + store (C/D layout: VGPR v -> row v / v+8, lane -> col) ----
    float alphaV = 0.0f;
    if (MODE == MODE_SOFT) alphaV = alphaPtr[0];

    #pragma unroll
    for (int f = 0; f < 4; ++f) {
        const int gf = f * 16 + lrow;
        #pragma unroll
        for (int v = 0; v < 8; ++v) {
            const int gm = row0 + wave * 16 + v + 8 * h;
            float r = acc[f][v];
            if (MODE == MODE_SOFT) {
                r = (r > alphaV) ? (r - alphaV)
                                 : ((r < -alphaV) ? (r + alphaV) : 0.0f);
            }
            if (MODE == MODE_ADDBX) {
                r += bx[(size_t)gm * FDIM + gf];
            }
            out[(size_t)gm * FDIM + gf] = (TO)r;
        }
    }
}

// ---------------------------------------------------------------------------
// Host-side orchestration.
//   y0 = 0  =>  y1 = Bx exactly (soft(L^T*0) = 0).  So: compute Bx once,
//   alias it as y1, then run 7 full steps; last step writes d_out.
//   One-time: conv_C -> bf16, L -> bf16 + transposed bf16 (all L2-resident
//   thereafter: 64+32+32 = 128 MB < 192 MB L2).
// ---------------------------------------------------------------------------
extern "C" void kernel_launch(void* const* d_in, const int* in_sizes, int n_in,
                              void* d_out, int out_size, void* d_ws, size_t ws_size,
                              hipStream_t stream)
{
    (void)in_sizes; (void)n_in; (void)out_size; (void)ws_size;

    const float* x_c    = (const float*)d_in[0];   // [B,N,F]
    const float* L      = (const float*)d_in[1];   // [N,E]
    const float* conv_B = (const float*)d_in[2];   // [B,N,N]
    const float* conv_C = (const float*)d_in[3];   // [B,N,N]
    const float* alpha  = (const float*)d_in[4];   // [1]

    float* yout = (float*)d_out;                   // [B,N,F]

    // Workspace carve-up
    const size_t ynf = (size_t)BATCH * NDIM * FDIM;   // 1,048,576
    const size_t zef = (size_t)BATCH * EDIM * FDIM;   // 4,194,304
    const size_t lne = (size_t)NDIM * EDIM;           // 16,777,216
    const size_t cnn = (size_t)BATCH * NDIM * NDIM;   // 33,554,432

    char* p = (char*)d_ws;
    float*  bxBuf   = (float*)p;   p += ynf * sizeof(float);   //  4 MB
    float*  lzBuf   = (float*)p;   p += ynf * sizeof(float);   //  4 MB
    float*  yA      = (float*)p;   p += ynf * sizeof(float);   //  4 MB
    float*  yB      = (float*)p;   p += ynf * sizeof(float);   //  4 MB
    __bf16* zBuf    = (__bf16*)p;  p += zef * sizeof(__bf16);  //  8 MB
    __bf16* Lbf     = (__bf16*)p;  p += lne * sizeof(__bf16);  // 32 MB
    __bf16* LTbf    = (__bf16*)p;  p += lne * sizeof(__bf16);  // 32 MB
    __bf16* convCbf = (__bf16*)p;                              // 64 MB

    const dim3 blk(256);
    const dim3 gridN(NDIM / MTILE, BATCH);   // 16 x 8
    const dim3 gridE(EDIM / MTILE, BATCH);   // 64 x 8

    // ---- one-time precompute: bf16 copies of the reused matrices ----
    cvt_f32_to_bf16<<<4096, blk, 0, stream>>>(conv_C, convCbf, cnn / 4);
    cvt_transpose_L<<<dim3(EDIM / 32, NDIM / 32), blk, 0, stream>>>(L, Lbf, LTbf);

    // ---- Bx = conv_B @ x_c   (also serves as y after step 1) ----
    gemm_wmma<float, float, float, MODE_PLAIN><<<gridN, blk, 0, stream>>>(
        conv_B, NDIM, (long)NDIM * NDIM,
        x_c,    (long)NDIM * FDIM,
        bxBuf,  (long)NDIM * FDIM,
        nullptr, 0, nullptr,
        NDIM, NDIM);

    const float* yprev = bxBuf;

    // ---- 7 remaining recurrence steps ----
    for (int s = 0; s < NUM_HIDDEN - 1; ++s) {
        // z = soft_threshold(L^T @ y)   [B,E,F] bf16  (A = pre-transposed L)
        gemm_wmma<__bf16, float, __bf16, MODE_SOFT><<<gridE, blk, 0, stream>>>(
            LTbf, NDIM, 0L,
            yprev, (long)NDIM * FDIM,
            zBuf,  (long)EDIM * FDIM,
            nullptr, 0, alpha,
            EDIM, NDIM);

        // Lz = L @ z                    [B,N,F] f32
        gemm_wmma<__bf16, __bf16, float, MODE_PLAIN><<<gridN, blk, 0, stream>>>(
            Lbf, EDIM, 0L,
            zBuf,  (long)EDIM * FDIM,
            lzBuf, (long)NDIM * FDIM,
            nullptr, 0, nullptr,
            NDIM, EDIM);

        // y' = conv_C @ Lz + Bx         [B,N,F] f32
        float* ynext = (s == NUM_HIDDEN - 2) ? yout : ((s & 1) ? yB : yA);
        gemm_wmma<__bf16, float, float, MODE_ADDBX><<<gridN, blk, 0, stream>>>(
            convCbf, NDIM, (long)NDIM * NDIM,
            lzBuf, (long)NDIM * FDIM,
            ynext, (long)NDIM * FDIM,
            bxBuf, (long)NDIM * FDIM, nullptr,
            NDIM, NDIM);

        yprev = ynext;
    }
}
